// DeformableAttention_67345087201542
// MI455X (gfx1250) — compile-verified, matrix operand order checked by reading, not run on hardware
//
#include <hip/hip_runtime.h>
#include <hip/hip_bf16.h>

typedef __attribute__((ext_vector_type(16))) __bf16 v16bf;
typedef __attribute__((ext_vector_type(8)))  float  v8f;

#define NUM_HEADS 8
#define N_POINTS  9
#define HEAD_DIM  32

// ---------------------------------------------------------------------------
// fp32 -> bf16 (round-to-nearest-even) conversion
// ---------------------------------------------------------------------------
__global__ void cvt_f32_bf16(const float* __restrict__ in,
                             unsigned short* __restrict__ out, int n) {
    int i = blockIdx.x * blockDim.x + threadIdx.x;
    if (i < n) {
        unsigned int b = __float_as_uint(in[i]);
        unsigned int r = b + 0x7FFFu + ((b >> 16) & 1u);
        out[i] = (unsigned short)(r >> 16);
    }
}

// ---------------------------------------------------------------------------
// bf16 WMMA GEMM:  C[M,N] = A[M,K] * W[N,K]^T + bias[N]
//   EPI = 0 : plain bias add (f32 out, row-major MxN)
//   EPI = 1 : tanh(acc+bias)*4 (offset head)
//   EPI = 2 : k/v channel-interleaved store (N==512):
//             dst = m*512 + (col&255)*2 + (col>>8)  ->  kv[m, chan, {k,v}]
// Wave tile: 16(M) x (16*NT)(N). Block = 4 waves, each a distinct M tile.
// A-frag, this code's contiguous k order (16-bit A 16x32): lane<16 -> row M=lane, K=k0..k0+15
//                                         lane>=16 -> row M=lane-16, K=k0+16..k0+31
// B-frag symmetric on W rows (W row n is the K-vector of B column n).
// ---------------------------------------------------------------------------
template<int EPI, int NT>
__global__ __launch_bounds__(128)
void gemm_bf16_wmma(const __bf16* __restrict__ A,
                    const __bf16* __restrict__ W,
                    const float*  __restrict__ bias,
                    float* __restrict__ C,
                    int M, int N, int K) {
    const int wave = threadIdx.x >> 5;
    const int lane = threadIdx.x & 31;
    const int mt   = blockIdx.x * 4 + wave;       // 16-row M tile
    const int nt0  = blockIdx.y * NT;             // first 16-col N tile
    const int mrow = mt * 16 + (lane & 15);
    const int klo  = (lane >> 4) * 16;            // 0 or 16

    v8f acc[NT];
#pragma unroll
    for (int j = 0; j < NT; ++j) acc[j] = (v8f){};

    const __bf16* arow = A + (size_t)mrow * K;
    for (int k0 = 0; k0 < K; k0 += 32) {
        v16bf a = *(const v16bf*)(arow + k0 + klo);
        __builtin_prefetch(arow + k0 + 64, 0, 1);   // global_prefetch_b8
#pragma unroll
        for (int j = 0; j < NT; ++j) {
            int wrow = (nt0 + j) * 16 + (lane & 15);
            v16bf b = *(const v16bf*)(W + (size_t)wrow * K + k0 + klo);
            acc[j] = __builtin_amdgcn_wmma_f32_16x16x32_bf16(
                false, a, false, b, (short)0, acc[j], false, false);
        }
    }

    // C/D layout: VGPR i -> M = mt*16 + i + 8*(lane>>4), N = nt*16 + (lane&15)
    const int mbase = mt * 16 + (lane >> 4) * 8;
#pragma unroll
    for (int j = 0; j < NT; ++j) {
        const int col = (nt0 + j) * 16 + (lane & 15);
        const float bs = bias[col];
#pragma unroll
        for (int i = 0; i < 8; ++i) {
            float v = acc[j][i] + bs;
            if (EPI == 1) v = tanhf(v) * 4.0f;
            if (EPI == 2) {
                // interleaved k/v layout for b64 gathers in sampling
                C[(size_t)(mbase + i) * N + (size_t)(col & 255) * 2 + (col >> 8)] = v;
            } else {
                C[(size_t)(mbase + i) * N + col] = v;
            }
        }
    }
}

// ---------------------------------------------------------------------------
// Deformable sampling + attention. One wave per (b, h, n); lane = channel.
// q (M,256) f32, offsets (M,144) f32 (already tanh*4),
// kv interleaved (M, 256, 2) f32: [..., chan, 0]=k, [..., chan, 1]=v.
// Writes attn output as bf16 (M,256) for the final WMMA GEMM.
// ---------------------------------------------------------------------------
__global__ __launch_bounds__(256)
void sample_attend(const float* __restrict__ qbuf,
                   const float* __restrict__ offbuf,
                   const float* __restrict__ kvbuf,
                   const int* __restrict__ hptr,
                   const int* __restrict__ wptr,
                   unsigned short* __restrict__ attn_bf,
                   int M /* = B*N */) {
    const int Himg = *hptr;                 // 40
    const int Wimg = *wptr;                 // 160
    const int Nsp  = Himg * Wimg;           // 6400

    const int waveId = blockIdx.x * (blockDim.x >> 5) + (threadIdx.x >> 5);
    const int lane   = threadIdx.x & 31;    // channel c in [0,32)
    const int m      = waveId >> 3;         // row in [0, M): m = b*Nsp + n
    const int hh     = waveId & 7;          // head
    if (m >= M) return;

    const int b = m / Nsp;
    const int n = m - b * Nsp;
    const float px = (float)(n % Wimg);
    const float py = (float)(n / Wimg);

    const int chan = hh * HEAD_DIM + lane;
    const float qc = qbuf[(size_t)m * 256 + chan];

    // one predicated lane-parallel load of the 18 offsets; broadcast via shfl
    const float* offp = offbuf + (size_t)m * (NUM_HEADS * N_POINTS * 2)
                               + hh * (N_POINTS * 2);
    const float myoff = (lane < 2 * N_POINTS) ? offp[lane] : 0.0f;

    const float2* kvb = (const float2*)(kvbuf + (size_t)b * Nsp * 512);

    float logits[N_POINTS];
    float sv[N_POINTS];

#pragma unroll
    for (int p = 0; p < N_POINTS; ++p) {
        const float sx = px + __shfl(myoff, 2 * p + 0, 32);
        const float sy = py + __shfl(myoff, 2 * p + 1, 32);
        const float x0 = floorf(sx), y0 = floorf(sy);
        const float wx1 = sx - x0, wx0 = 1.0f - wx1;
        const float wy1 = sy - y0, wy0 = 1.0f - wy1;

        float kc = 0.0f, vc = 0.0f;
#pragma unroll
        for (int cy = 0; cy < 2; ++cy) {
#pragma unroll
            for (int cx = 0; cx < 2; ++cx) {
                const float xi = x0 + (float)cx;
                const float yi = y0 + (float)cy;
                const bool valid = (xi >= 0.0f) & (xi <= (float)(Wimg - 1)) &
                                   (yi >= 0.0f) & (yi <= (float)(Himg - 1));
                const float wgt = (cx ? wx1 : wx0) * (cy ? wy1 : wy0) *
                                  (valid ? 1.0f : 0.0f);
                const int xic = min(max((int)xi, 0), Wimg - 1);
                const int yic = min(max((int)yi, 0), Himg - 1);
                // single b64 gather: (k, v) pair for this channel
                const float2 kv2 = kvb[(size_t)(yic * Wimg + xic) * 256 + chan];
                kc += wgt * kv2.x;
                vc += wgt * kv2.y;
            }
        }
        // cross-lane (channel) dot product reduction, wave32 butterfly
        float l = qc * kc;
#pragma unroll
        for (int o = 16; o; o >>= 1) l += __shfl_xor(l, o, 32);
        logits[p] = l * 0.17677669529663687f;   // 1/sqrt(32)
        sv[p] = vc;
    }

    float mmax = logits[0];
#pragma unroll
    for (int p = 1; p < N_POINTS; ++p) mmax = fmaxf(mmax, logits[p]);
    float sum = 0.0f, oc = 0.0f;
#pragma unroll
    for (int p = 0; p < N_POINTS; ++p) {
        const float e = __expf(logits[p] - mmax);
        sum += e;
        oc  += e * sv[p];
    }
    oc /= sum;

    unsigned int bb = __float_as_uint(oc);
    unsigned int r  = bb + 0x7FFFu + ((bb >> 16) & 1u);
    attn_bf[(size_t)m * 256 + chan] = (unsigned short)(r >> 16);
}

// ---------------------------------------------------------------------------
// Host orchestration
// ---------------------------------------------------------------------------
extern "C" void kernel_launch(void* const* d_in, const int* in_sizes, int n_in,
                              void* d_out, int out_size, void* d_ws, size_t ws_size,
                              hipStream_t stream) {
    const float* x    = (const float*)d_in[0];
    const float* Wq   = (const float*)d_in[1];
    const float* bq   = (const float*)d_in[2];
    const float* Woff = (const float*)d_in[3];
    const float* boff = (const float*)d_in[4];
    const float* Wkv  = (const float*)d_in[5];
    const float* bkv  = (const float*)d_in[6];
    const float* Wout = (const float*)d_in[7];
    const float* bout = (const float*)d_in[8];
    const int*   hptr = (const int*)d_in[9];
    const int*   wptr = (const int*)d_in[10];

    const int D    = 256;
    const int M    = in_sizes[0] / D;    // B*N = 25600
    const int Noff = in_sizes[3] / D;    // 144
    const int Nkv  = in_sizes[5] / D;    // 512

    // workspace carve-up (256B aligned slabs)
    char* p = (char*)d_ws;
    auto carve = [&](size_t bytes) {
        char* r = p;
        p += (bytes + 255) & ~(size_t)255;
        return r;
    };
    unsigned short* x_bf    = (unsigned short*)carve((size_t)M * D * 2);
    unsigned short* wq_bf   = (unsigned short*)carve((size_t)in_sizes[1] * 2);
    unsigned short* woff_bf = (unsigned short*)carve((size_t)in_sizes[3] * 2);
    unsigned short* wkv_bf  = (unsigned short*)carve((size_t)in_sizes[5] * 2);
    unsigned short* wout_bf = (unsigned short*)carve((size_t)in_sizes[7] * 2);
    float*          qbuf    = (float*)carve((size_t)M * D * 4);
    float*          offbuf  = (float*)carve((size_t)M * Noff * 4);
    float*          kvbuf   = (float*)carve((size_t)M * Nkv * 4);
    unsigned short* attn_bf = (unsigned short*)carve((size_t)M * D * 2);

    // 1) fp32 -> bf16 conversions
    {
        int n;
        n = M * D;        cvt_f32_bf16<<<(n + 255) / 256, 256, 0, stream>>>(x,    x_bf,    n);
        n = in_sizes[1];  cvt_f32_bf16<<<(n + 255) / 256, 256, 0, stream>>>(Wq,   wq_bf,   n);
        n = in_sizes[3];  cvt_f32_bf16<<<(n + 255) / 256, 256, 0, stream>>>(Woff, woff_bf, n);
        n = in_sizes[5];  cvt_f32_bf16<<<(n + 255) / 256, 256, 0, stream>>>(Wkv,  wkv_bf,  n);
        n = in_sizes[7];  cvt_f32_bf16<<<(n + 255) / 256, 256, 0, stream>>>(Wout, wout_bf, n);
    }

    const int mblocks = M / 64;   // 4 waves/block, 16 rows/wave

    // 2) q = x @ Wq^T + bq            (M x 256)
    gemm_bf16_wmma<0, 4><<<dim3(mblocks, D / 64), 128, 0, stream>>>(
        (const __bf16*)x_bf, (const __bf16*)wq_bf, bq, qbuf, M, D, D);

    // 3) offsets = tanh(x @ Woff^T + boff) * 4   (M x 144)
    gemm_bf16_wmma<1, 3><<<dim3(mblocks, Noff / 48), 128, 0, stream>>>(
        (const __bf16*)x_bf, (const __bf16*)woff_bf, boff, offbuf, M, Noff, D);

    // 4) kv = x @ Wkv^T + bkv          (M x 512), k/v channel-interleaved
    gemm_bf16_wmma<2, 4><<<dim3(mblocks, Nkv / 64), 128, 0, stream>>>(
        (const __bf16*)x_bf, (const __bf16*)wkv_bf, bkv, kvbuf, M, Nkv, D);

    // 5) deformable sampling + attention: M*8 waves, 8 waves/block
    sample_attend<<<M, 256, 0, stream>>>(qbuf, offbuf, kvbuf, hptr, wptr,
                                         attn_bf, M);

    // 6) out = attn @ Wout^T + bout    (M x 256) -> d_out (fp32)
    gemm_bf16_wmma<0, 4><<<dim3(mblocks, D / 64), 128, 0, stream>>>(
        (const __bf16*)attn_bf, (const __bf16*)wout_bf, bout, (float*)d_out,
        M, D, D);
}
